// SimpleGenerativeLALRNN_36498632081935
// MI455X (gfx1250) — compile-verified, matrix operand order checked by reading, not run on hardware
//
#include <hip/hip_runtime.h>
#include <hip/hip_bf16.h>

typedef _Float16 half_t;
typedef _Float16 v16h __attribute__((ext_vector_type(16)));
typedef _Float16 h8   __attribute__((ext_vector_type(8)));
typedef float    v8f  __attribute__((ext_vector_type(8)));
typedef float    f4   __attribute__((ext_vector_type(4)));

#define B_    4096
#define T_    128
#define NL_   256
#define NS_   64
#define NO_   32
#define V_    64
#define XS_STRIDE 264   // halfs per LDS row; 528B => rows rotate 4 banks => conflict-free b128 frag loads

// ---------------- prep: f32 -> f16 weight conversion (row-major, layout-preserving) ----------------
__global__ void cvt_f32_to_f16_kernel(const float* __restrict__ in, half_t* __restrict__ out, int n) {
  int i = blockIdx.x * blockDim.x + threadIdx.x;
  if (i < n) out[i] = (half_t)in[i];
}

// A/B fragment loader for v_wmma_f32_16x16x32_f16.
// 16-bit 16x32 A layout: lanes 0-15 hold row M=lane with K = base+{0..7} and base+{16..23};
// lanes 16-31 hold M=lane-16 with K = base+{8..15} and base+{24..31}.  (B mirrored with N<->M.)
// kbase already includes the per-lane-half +8 adjustment.
__device__ __forceinline__ v16h load_frag_lds(const half_t* base, int row, int kbase) {
  const half_t* p = base + row * XS_STRIDE + kbase;
  v16h a;
  *((h8*)&a)       = *((const h8*)p);
  *(((h8*)&a) + 1) = *((const h8*)(p + 16));
  return a;
}

__device__ __forceinline__ v16h load_frag_glb(const half_t* rowptr, int kbase) {
  v16h a;
  *((h8*)&a)       = *((const h8*)(rowptr + kbase));
  *(((h8*)&a) + 1) = *((const h8*)(rowptr + kbase + 16));
  return a;
}

__device__ __forceinline__ float fast_tanh(float v) {
  // tanh(v) = 1 - 2/(exp(2v)+1); saturates correctly at +-1 for large |v|
  float e = __expf(2.0f * v);
  return 1.0f - 2.0f * __builtin_amdgcn_rcpf(e + 1.0f);
}

// ---------------- persistent per-batch-slice kernel: all T steps, no global sync ----------------
__global__ __launch_bounds__(128)
void lalr_persistent_kernel(const float* __restrict__ latent0,
                            const half_t* __restrict__ W1h,   // [NS][NL][NL] f16
                            const half_t* __restrict__ W2h,   // [NS][NO][NL] f16
                            const float* __restrict__ b1,     // [NS][NL]
                            const float* __restrict__ b2,     // [NS][NO]
                            const int* __restrict__ state_seq,// [T]
                            const int* __restrict__ out_idx,  // [NS][NO]
                            float* __restrict__ out) {        // [B][T][V]
  __shared__ half_t xs[2][16][XS_STRIDE];   // double-buffered latent (f16)
  __shared__ float  logits[16][33];         // [row][NO], padded
  __shared__ float  ybuf[16][V_];           // scatter staging

  const int tid   = threadIdx.x;
  const int wave  = tid >> 5;
  const int lane  = tid & 31;
  const int lhalf = lane & 15;
  const int hi    = lane >> 4;              // 0 or 1
  const int mbase = hi ? 8 : 0;             // C-tile M offset for this lane half
  const int row0  = blockIdx.x * 16;

  // ---- initial load: latent0 (f32) -> xs[0] (f16), coalesced ----
  {
    int r  = tid >> 3;              // 16 rows, 8 threads per row
    int c0 = (tid & 7) * 32;        // 32 cols each
    const float* src = latent0 + (size_t)(row0 + r) * NL_ + c0;
    alignas(16) half_t tmp[32];
#pragma unroll
    for (int i = 0; i < 8; ++i) {
      f4 v = *((const f4*)(src + i * 4));
#pragma unroll
      for (int k = 0; k < 4; ++k) tmp[i * 4 + k] = (half_t)v[k];
    }
#pragma unroll
    for (int i = 0; i < 4; ++i)
      *((h8*)&xs[0][r][c0 + i * 8]) = *((const h8*)&tmp[i * 8]);
  }
  __syncthreads();

  int cur = 0;
  for (int t = 0; t < T_; ++t) {
    const int s   = state_seq[t];         // uniform -> s_load
    const int nxt = cur ^ 1;
    const int kb  = hi ? 8 : 0;

    // ---- Phase A: x_new = tanh(x @ W1[s]^T + b1[s]); wave handles N cols [wave*64, wave*64+64) ----
    {
      v8f acc[4] = {};
#pragma unroll
      for (int kt = 0; kt < 8; ++kt) {
        v16h a = load_frag_lds(&xs[cur][0][0], lhalf, kt * 32 + kb);
#pragma unroll
        for (int nt = 0; nt < 4; ++nt) {
          int n = wave * 64 + nt * 16 + lhalf;
          const half_t* wrow = W1h + ((size_t)(s * NL_ + n)) * NL_;
          v16h b = load_frag_glb(wrow, kt * 32 + kb);
          acc[nt] = __builtin_amdgcn_wmma_f32_16x16x32_f16(
              false, a, false, b, (short)0, acc[nt], false, false);
        }
      }
#pragma unroll
      for (int nt = 0; nt < 4; ++nt) {
        int col = wave * 64 + nt * 16 + lhalf;
        float bb = b1[s * NL_ + col];
#pragma unroll
        for (int j = 0; j < 8; ++j) {
          // C layout: VGPR j -> M = mbase + j, N = lhalf (within tile)
          float v = fast_tanh(acc[nt][j] + bb);
          xs[nxt][mbase + j][col] = (half_t)v;
        }
      }
    }
    __syncthreads();  // B1: x_new complete; also orders prev-iter output copy vs ybuf reuse

    // ---- Phase B: logits = x_new @ W2[s]^T + b2[s] (waves 0,1); all threads zero ybuf ----
    {
      int zb = tid * 8;
#pragma unroll
      for (int i = 0; i < 8; ++i) ((float*)ybuf)[zb + i] = 0.0f;

      if (wave < 2) {
        v8f acc2 = {};
#pragma unroll
        for (int kt = 0; kt < 8; ++kt) {
          v16h a = load_frag_lds(&xs[nxt][0][0], lhalf, kt * 32 + kb);
          int n2 = wave * 16 + lhalf;
          const half_t* wrow = W2h + ((size_t)(s * NO_ + n2)) * NL_;
          v16h b = load_frag_glb(wrow, kt * 32 + kb);
          acc2 = __builtin_amdgcn_wmma_f32_16x16x32_f16(
              false, a, false, b, (short)0, acc2, false, false);
        }
        float bb = b2[s * NO_ + wave * 16 + lhalf];
#pragma unroll
        for (int j = 0; j < 8; ++j)
          logits[mbase + j][wave * 16 + lhalf] = acc2[j] + bb;
      }
    }
    __syncthreads();  // B2

    // ---- Phase C: per-row softmax + IndicesExpansion scatter (16 lanes) ----
    if (tid < 16) {
      int r = tid;
      float mx = -3.402823466e38f;
#pragma unroll
      for (int j = 0; j < NO_; ++j) mx = fmaxf(mx, logits[r][j]);
      float sum = 0.0f;
#pragma unroll
      for (int j = 0; j < NO_; ++j) {
        float e = __expf(logits[r][j] - mx);
        logits[r][j] = e;
        sum += e;
      }
      float inv = __builtin_amdgcn_rcpf(sum);
      for (int j = 0; j < NO_; ++j) {           // ascending j => last index wins (scatter .set)
        int ix = out_idx[s * NO_ + j];
        ybuf[r][ix] = logits[r][j] * inv;
      }
    }
    __syncthreads();  // B3

    // ---- Phase D: copy ybuf -> out[row, t, :], coalesced f4 stores ----
    {
      int r  = tid >> 3;
      int c0 = (tid & 7) * 8;
      float* dst = out + (size_t)(row0 + r) * (T_ * V_) + (size_t)t * V_ + c0;
      f4 v0 = *((const f4*)&ybuf[r][c0]);
      f4 v1 = *((const f4*)&ybuf[r][c0 + 4]);
      *((f4*)dst)       = v0;
      *((f4*)(dst + 4)) = v1;
    }
    cur = nxt;
  }
}

extern "C" void kernel_launch(void* const* d_in, const int* in_sizes, int n_in,
                              void* d_out, int out_size, void* d_ws, size_t ws_size,
                              hipStream_t stream) {
  const float* latent0   = (const float*)d_in[0];
  const float* W1        = (const float*)d_in[1];
  const float* b1        = (const float*)d_in[2];
  const float* W2        = (const float*)d_in[3];
  const float* b2        = (const float*)d_in[4];
  const int*   state_seq = (const int*)d_in[5];
  const int*   out_idx   = (const int*)d_in[6];
  float* out = (float*)d_out;

  // workspace layout: f16 W1 (8 MiB) | f16 W2 (1 MiB)
  half_t* W1h = (half_t*)d_ws;
  half_t* W2h = (half_t*)((char*)d_ws + (size_t)NS_ * NL_ * NL_ * sizeof(half_t));

  {
    int n1 = NS_ * NL_ * NL_;
    cvt_f32_to_f16_kernel<<<(n1 + 255) / 256, 256, 0, stream>>>(W1, W1h, n1);
    int n2 = NS_ * NO_ * NL_;
    cvt_f32_to_f16_kernel<<<(n2 + 255) / 256, 256, 0, stream>>>(W2, W2h, n2);
  }

  lalr_persistent_kernel<<<B_ / 16, 128, 0, stream>>>(
      latent0, W1h, W2h, b1, b2, state_seq, out_idx, out);
}
